// Distance_Loss_57200374448345
// MI455X (gfx1250) — compile-verified, matrix operand order checked by reading, not run on hardware
//
#include <hip/hip_runtime.h>
#include <hip/hip_bf16.h>
#include <hip/hip_fp16.h>

// ---------------- problem constants (match reference) ----------------
#define BIGF    100000000.0f
#define EPS_Wf  0.001f
#define SMOOTHf 1e-06f
#define ALPHAf  0.6f
#define BETAf   1.0f

static constexpr int B_  = 16;
static constexpr int H_  = 512;
static constexpr int W_  = 512;
static constexpr int HW_ = H_ * W_;            // 262144 = 2^18
static constexpr int N_  = B_ * HW_;           // 4194304
static constexpr int NB_ = 1024;               // partial-reduce blocks

typedef __attribute__((ext_vector_type(16))) _Float16 v16h;
typedef __attribute__((ext_vector_type(8)))  float    v8f;

// ---------------------------------------------------------------------
// Wave-level sum reduction on the matrix pipe:
//   each lane supplies 16 fp32 values; they are split into f16 hi/lo and
//   summed by D = ones(16x32) x B(32x16) + C with fp32 accumulation
//   (2 x v_wmma_f32_16x16x32_f16).  Every row of D equals the 16 column
//   sums, so d[0] across lanes holds all column sums; a 4-step butterfly
//   finishes.  Exact to ~2^-22 relative.  Requires full EXEC.
// ---------------------------------------------------------------------
__device__ __forceinline__ float wmma_wave_sum(const float* vals) {
  v16h ones, bh, bl;
#pragma unroll
  for (int i = 0; i < 16; ++i) {
    ones[i] = (_Float16)1.0f;
    float v = vals[i];
    _Float16 h = (_Float16)v;
    bh[i] = h;
    bl[i] = (_Float16)(v - (float)h);
  }
  v8f c = {};
  // 8 args: (neg_a, A, neg_b, B, c_mod, C, reuse_a, reuse_b)
  c = __builtin_amdgcn_wmma_f32_16x16x32_f16(false, ones, false, bh, (short)0, c, false, false);
  c = __builtin_amdgcn_wmma_f32_16x16x32_f16(false, ones, false, bl, (short)0, c, false, false);
  float s = c[0];                 // lane L: column-sum(L % 16)
  s += __shfl_xor(s, 8, 32);
  s += __shfl_xor(s, 4, 32);
  s += __shfl_xor(s, 2, 32);
  s += __shfl_xor(s, 1, 32);
  return s;                       // total of all 512 inputs, uniform per lane
}

// ---------------------------------------------------------------------
// Kernel 1: vertical two-way scan.  One thread per (b, w) column.
// fwd pass stores running distance; bwd pass combines:
//   g = min(min(fwd,bwd)^2, BIG).
// Also zero-inits the per-batch atomic-max slots.
// ---------------------------------------------------------------------
__global__ void edt_vertical(const float* __restrict__ tg,
                             float* __restrict__ g,
                             unsigned int* __restrict__ vbits) {
  int tid = blockIdx.x * blockDim.x + threadIdx.x;
  if (tid < B_) vbits[tid] = 0u;
  if (tid >= B_ * W_) return;
  int b = tid / W_, w = tid - b * W_;
  const float* tcol = tg + (size_t)b * HW_ + w;
  float*       gcol = g  + (size_t)b * HW_ + w;
  float c = BIGF;
  for (int h = 0; h < H_; ++h) {
    float t = tcol[(size_t)h * W_];
    c = (t == 0.0f) ? 0.0f : (c + 1.0f);
    gcol[(size_t)h * W_] = c;
  }
  c = BIGF;
  for (int h = H_ - 1; h >= 0; --h) {
    float t = tcol[(size_t)h * W_];
    c = (t == 0.0f) ? 0.0f : (c + 1.0f);
    float d1 = fminf(gcol[(size_t)h * W_], c);
    gcol[(size_t)h * W_] = fminf(d1 * d1, BIGF);
  }
}

// ---------------------------------------------------------------------
// Kernel 2: horizontal parabola pass (exact, brute force like reference),
// sqrt, in-place overwrite of g with the final distance, per-batch max
// via uint atomicMax (distances are >= 0 so float bits are monotone),
// and the rows[b, :] = dist(h == b) slice.
// One 256-thread block per (b, h) row.  The row is staged in LDS via the
// CDNA5 async memory->LDS path (global_load_async_to_lds_b128, tracked by
// ASYNCcnt): 128 lanes x 16B = 2048B, no VGPR bounce.
// ---------------------------------------------------------------------
__global__ void edt_horizontal(float* __restrict__ g,
                               unsigned int* __restrict__ vbits,
                               float* __restrict__ rows) {
  __shared__ __align__(16) float srow[W_];
  __shared__ float smax[256];
  int row = blockIdx.x;               // b*H + h
  int b = row / H_, h = row - b * H_;
  float* gr = g + (size_t)row * W_;
  int t = threadIdx.x;

  // --- async LDS fill: threads 0..127 each copy 16 bytes memory->LDS ---
  if (t < 128) {
    unsigned lds_addr = (unsigned)(uintptr_t)(&srow[t * 4]);   // LDS byte addr = addr[31:0]
    const float* gaddr = gr + t * 4;
    asm volatile("global_load_async_to_lds_b128 %0, %1, off"
                 :: "v"(lds_addr), "v"(gaddr)
                 : "memory");
  }
  asm volatile("s_wait_asynccnt 0" ::: "memory");
  __syncthreads();

  float dmax = 0.0f;
#pragma unroll
  for (int yi = 0; yi < 2; ++yi) {
    int y = t + yi * 256;
    float fy = (float)y;
    float best = BIGF;
#pragma unroll 8
    for (int j = 0; j < W_; ++j) {
      float dy = fy - (float)j;
      best = fminf(best, srow[j] + dy * dy);
    }
    float dist = sqrtf(best);
    gr[y] = dist;                     // safe: row already in LDS
    dmax = fmaxf(dmax, dist);
    if (h == b) rows[b * W_ + y] = dist;
  }
  smax[t] = dmax;
  __syncthreads();
  for (int s2 = 128; s2 > 0; s2 >>= 1) {
    if (t < s2) smax[t] = fmaxf(smax[t], smax[t + s2]);
    __syncthreads();
  }
  if (t == 0) atomicMax(&vbits[b], __float_as_uint(smax[0]));
}

// ---------------------------------------------------------------------
// Kernel 3: fused elementwise loss + partial reductions.
//   q0 = sum weight*(in-tg)^2,  q1 = sum sigmoid(in),
//   q2 = sum tg,                q3 = sum sigmoid(in)*tg
// Grid-stride; per-wave WMMA reduction; deterministic per-block partials.
// ---------------------------------------------------------------------
__global__ void loss_partials(const float* __restrict__ in,
                              const float* __restrict__ tg,
                              const float* __restrict__ dist,
                              const unsigned int* __restrict__ vbits,
                              const float* __restrict__ rows,
                              float* __restrict__ partials) {
  float a0 = 0.f, a1 = 0.f, a2 = 0.f, a3 = 0.f;
  int stride = gridDim.x * blockDim.x;
  for (int idx = blockIdx.x * blockDim.x + threadIdx.x; idx < N_; idx += stride) {
    __builtin_prefetch(&in[idx + stride], 0, 1);     // global_prefetch_b8
    __builtin_prefetch(&tg[idx + stride], 0, 1);
    int b = idx >> 18;                // /HW_
    int y = idx & (W_ - 1);
    float x = in[idx], t = tg[idx];
    float d = dist[idx];
    float m = (d != 0.0f) ? 1.0f : 0.0f;
    float vb = __uint_as_float(vbits[b]);
    float wgt = m * (vb - rows[b * W_ + y]) + EPS_Wf;
    float diff = x - t;
    a0 += wgt * diff * diff;
    float p = 1.0f / (1.0f + expf(-x));
    a1 += p;
    a2 += t;
    a3 += p * t;
  }
  float vals[16];
#pragma unroll
  for (int i = 1; i < 16; ++i) vals[i] = 0.0f;
  __shared__ float swsum[8 * 4];
  int lane = threadIdx.x & 31, wid = threadIdx.x >> 5;
  vals[0] = a0; float s0 = wmma_wave_sum(vals);
  vals[0] = a1; float s1 = wmma_wave_sum(vals);
  vals[0] = a2; float s2 = wmma_wave_sum(vals);
  vals[0] = a3; float s3 = wmma_wave_sum(vals);
  if (lane == 0) {
    swsum[wid * 4 + 0] = s0; swsum[wid * 4 + 1] = s1;
    swsum[wid * 4 + 2] = s2; swsum[wid * 4 + 3] = s3;
  }
  __syncthreads();
  if (threadIdx.x < 4) {
    float s = 0.f;
    for (int wv = 0; wv < 8; ++wv) s += swsum[wv * 4 + threadIdx.x];
    partials[threadIdx.x * NB_ + blockIdx.x] = s;
  }
}

// ---------------------------------------------------------------------
// Kernel 4: final reduction (1 block, 4 waves; wave q reduces quantity q's
// 1024 partials: 32 per lane pre-added in pairs to 16, then WMMA), and
// emits the two scalars.
// ---------------------------------------------------------------------
__global__ void finalize_loss(const float* __restrict__ partials,
                              float* __restrict__ out) {
  __shared__ float ssum[4];
  int lane = threadIdx.x & 31, q = threadIdx.x >> 5;
  const float* p = partials + q * NB_;
  float vals[16];
#pragma unroll
  for (int i = 0; i < 16; ++i)
    vals[i] = p[lane * 32 + 2 * i] + p[lane * 32 + 2 * i + 1];
  float s = wmma_wave_sum(vals);
  if (lane == 0) ssum[q] = s;
  __syncthreads();
  if (threadIdx.x == 0) {
    float wsum = ssum[0], psum = ssum[1], tsum = ssum[2], ptsum = ssum[3];
    out[0] = ALPHAf * (wsum / (float)N_);
    float dice = 1.0f - (2.0f * ptsum + SMOOTHf) / (psum + tsum + SMOOTHf);
    out[1] = BETAf * dice;
  }
}

// ---------------------------------------------------------------------
extern "C" void kernel_launch(void* const* d_in, const int* in_sizes, int n_in,
                              void* d_out, int out_size, void* d_ws, size_t ws_size,
                              hipStream_t stream) {
  const float* in = (const float*)d_in[0];   // inputs  (16,1,512,512) f32
  const float* tg = (const float*)d_in[1];   // targets (16,1,512,512) f32
  float* ws = (float*)d_ws;

  // workspace layout (floats): g/dist[N_], vbits[16], rows[B*W], partials[4*NB]
  float*        g        = ws;
  unsigned int* vbits    = (unsigned int*)(ws + N_);
  float*        rows     = ws + N_ + 16;
  float*        partials = ws + N_ + 16 + B_ * W_;
  float*        out      = (float*)d_out;

  edt_vertical  <<<(B_ * W_) / 256, 256, 0, stream>>>(tg, g, vbits);
  edt_horizontal<<<B_ * H_,        256, 0, stream>>>(g, vbits, rows);
  loss_partials <<<NB_,            256, 0, stream>>>(in, tg, g, vbits, rows, partials);
  finalize_loss <<<1,              128, 0, stream>>>(partials, out);
}
